// RNN_17051020165641
// MI455X (gfx1250) — matrix operand order repaired and checked by reading, not compiled, not run on hardware
//
#include <hip/hip_runtime.h>
#include <hip/hip_bf16.h>
#include <math.h>

// Problem constants (from the reference):
static constexpr int Vv = 32000;   // vocab
static constexpr int Ee = 256;     // embed dim
static constexpr int Hh = 512;     // hidden
static constexpr int Bb = 16;      // batch
static constexpr int Tt = 256;     // seq len
static constexpr int BT = Bb * Tt; // 4096 rows

typedef __attribute__((ext_vector_type(16))) _Float16 v16h;
typedef __attribute__((ext_vector_type(8)))  _Float16 v8h;
typedef __attribute__((ext_vector_type(8)))  float    v8f;

union H16 { v16h v; struct { v8h lo, hi; } p; };

// ---------------------------------------------------------------------------
// Embedding gather + f32->f16, into step-major layout [T*B, E]
// row r = t*16 + b  (so each timestep's 16 batch rows are contiguous)
// ---------------------------------------------------------------------------
__global__ void gather_embed_f16(const int* __restrict__ X,
                                 const float* __restrict__ emb,
                                 _Float16* __restrict__ Xe) {
  const int r = blockIdx.x;           // t*16 + b
  const int b = r & 15;
  const int t = r >> 4;
  const int tok = X[b * Tt + t];
  Xe[(size_t)r * Ee + threadIdx.x] = (_Float16)emb[(size_t)tok * Ee + threadIdx.x];
}

__global__ void cvt_f32_to_f16(const float* __restrict__ in,
                               _Float16* __restrict__ out, int n) {
  const int i = blockIdx.x * blockDim.x + threadIdx.x;
  if (i < n) out[i] = (_Float16)in[i];
}

// ---------------------------------------------------------------------------
// Tiled WMMA GEMM:  C[M,N] (f32) = A[M,K] (f16, row-major) x W[N,K]^T (f16)
//                   + bias0[n] + bias1[n]
// Block = 256 threads = 8 waves arranged 2(M) x 4(N); each wave: 16x64 strip.
// remap==1: rows are t*16+b, store to logits layout (b*T+t)*N + n with NT hint.
// ---------------------------------------------------------------------------
__global__ void __launch_bounds__(256)
wmma_gemm_abt(const _Float16* __restrict__ A, const _Float16* __restrict__ Bw,
              const float* __restrict__ bias0, const float* __restrict__ bias1,
              float* __restrict__ C, int M, int N, int K, int remap) {
  const int lane = threadIdx.x & 31;
  const int wave = threadIdx.x >> 5;                  // 0..7
  const int m0 = blockIdx.y * 32 + (wave >> 2) * 16;  // 2 M-tiles per block
  const int n0 = blockIdx.x * 256 + (wave & 3) * 64;  // 4 N-waves x 4 N-tiles
  const int l16 = lane & 15;
  const int hi  = (lane >> 4) & 1;
  (void)M;

  v8f acc0 = {}, acc1 = {}, acc2 = {}, acc3 = {};

  // A fragment (16x32 f16): lanes 0-15 -> K {k..k+15},
  //                         lanes 16-31 -> shifted by +16.
  const _Float16* arow = A  + (size_t)(m0 + l16) * K + hi * 16;
  // B fragment (32x16 f16): lane holds 16 contiguous K-values of one W row.
  const _Float16* br0  = Bw + (size_t)(n0 + l16) * K + hi * 16;
  const _Float16* br1  = br0 + (size_t)16 * K;
  const _Float16* br2  = br1 + (size_t)16 * K;
  const _Float16* br3  = br2 + (size_t)16 * K;

  for (int k = 0; k < K; k += 32) {
    H16 a, b0, b1, b2, b3;
    a.p.lo  = *(const v8h*)(arow + k);
    a.p.hi  = *(const v8h*)(arow + k + 8);
    b0.p.lo = *(const v8h*)(br0 + k);  b0.p.hi = *(const v8h*)(br0 + k + 8);
    b1.p.lo = *(const v8h*)(br1 + k);  b1.p.hi = *(const v8h*)(br1 + k + 8);
    b2.p.lo = *(const v8h*)(br2 + k);  b2.p.hi = *(const v8h*)(br2 + k + 8);
    b3.p.lo = *(const v8h*)(br3 + k);  b3.p.hi = *(const v8h*)(br3 + k + 8);
    acc0 = __builtin_amdgcn_wmma_f32_16x16x32_f16(false, a.v, false, b0.v, (short)0, acc0, false, false);
    acc1 = __builtin_amdgcn_wmma_f32_16x16x32_f16(false, a.v, false, b1.v, (short)0, acc1, false, false);
    acc2 = __builtin_amdgcn_wmma_f32_16x16x32_f16(false, a.v, false, b2.v, (short)0, acc2, false, false);
    acc3 = __builtin_amdgcn_wmma_f32_16x16x32_f16(false, a.v, false, b3.v, (short)0, acc3, false, false);
  }

  // C/D layout: VGPR r -> (lanes 0-15: M=r, lanes 16-31: M=r+8), N = lane%16.
  const int mb = m0 + hi * 8;
  v8f accs[4] = {acc0, acc1, acc2, acc3};
#pragma unroll
  for (int j = 0; j < 4; ++j) {
    const int n = n0 + j * 16 + l16;
    float bs = 0.0f;
    if (bias0) bs += bias0[n];
    if (bias1) bs += bias1[n];
#pragma unroll
    for (int r = 0; r < 8; ++r) {
      const int m = mb + r;
      const float val = accs[j][r] + bs;
      if (remap == 0) {
        C[(size_t)m * N + n] = val;
      } else {
        // row m = t*16+b  ->  logits[(b*T + t)*V + n], streaming (NT) store
        const size_t off = ((size_t)(m & 15) * Tt + (m >> 4)) * (size_t)N + n;
        __builtin_nontemporal_store(val, C + off);
      }
    }
  }
}

// ---------------------------------------------------------------------------
// Sequential Elman scan, one workgroup of 32 waves (1024 threads).
// Wave w owns output columns [16w, 16w+16); keeps the 16 W_hh B-fragments
// for those columns register-resident across all T steps. h is double-
// buffered f16 in LDS; one barrier per timestep.
//   h_t = tanh(xp_t + h_{t-1} @ W_hh^T)
// xp / outh are step-major [T*16, H]. out2f (nullable) gets fp32 [B,T,H];
// hT gets the final hidden state [B,H] in fp32.
// ---------------------------------------------------------------------------
__global__ void __launch_bounds__(1024)
rnn_scan(const _Float16* __restrict__ Whh, const float* __restrict__ xp,
         _Float16* __restrict__ outh, float* __restrict__ hT,
         float* __restrict__ out2f) {
  __shared__ _Float16 hbuf[2][Bb * Hh];   // 2 x 16KB
  const int lane = threadIdx.x & 31;
  const int wave = threadIdx.x >> 5;      // 0..31
  const int n0   = wave * 16;
  const int l16  = lane & 15;
  const int hi   = (lane >> 4) & 1;

  // Load resident W_hh B-fragments: lane covers W[n0+l16, kc*32 + 16*hi .. +15]
  H16 bw[16];
  {
    const _Float16* wrow = Whh + (size_t)(n0 + l16) * Hh + hi * 16;
#pragma unroll
    for (int kc = 0; kc < 16; ++kc) {
      bw[kc].p.lo = *(const v8h*)(wrow + kc * 32);
      bw[kc].p.hi = *(const v8h*)(wrow + kc * 32 + 8);
    }
  }

  // h_0 = 0
  for (int i = threadIdx.x; i < Bb * Hh; i += 1024) hbuf[0][i] = (_Float16)0.0f;
  __syncthreads();

  const int n  = n0 + l16;
  const int mb = hi * 8;
  for (int t = 0; t < Tt; ++t) {
    const _Float16* hc = hbuf[t & 1];
    _Float16*       hn = hbuf[(t + 1) & 1];

    v8f acc = {};
    const _Float16* ha = hc + l16 * Hh + hi * 16;
#pragma unroll
    for (int kc = 0; kc < 16; ++kc) {
      H16 a;
      a.p.lo = *(const v8h*)(ha + kc * 32);        // ds_load_b128
      a.p.hi = *(const v8h*)(ha + kc * 32 + 8);
      acc = __builtin_amdgcn_wmma_f32_16x16x32_f16(false, a.v, false, bw[kc].v,
                                                   (short)0, acc, false, false);
    }

#pragma unroll
    for (int r = 0; r < 8; ++r) {
      const int m = mb + r;                        // batch index b
      const float v = tanhf(acc[r] + xp[((size_t)t * Bb + m) * Hh + n]);
      const _Float16 hv = (_Float16)v;
      hn[m * Hh + n] = hv;                         // next-step h (LDS)
      outh[((size_t)t * Bb + m) * Hh + n] = hv;    // f16 layer output
      if (out2f) out2f[((size_t)m * Tt + t) * Hh + n] = v;  // [B,T,H] fp32
      if (t == Tt - 1) hT[m * Hh + n] = v;         // final hidden
    }
    __syncthreads();
  }
}

// ---------------------------------------------------------------------------
extern "C" void kernel_launch(void* const* d_in, const int* in_sizes, int n_in,
                              void* d_out, int out_size, void* d_ws, size_t ws_size,
                              hipStream_t stream) {
  (void)in_sizes; (void)n_in; (void)out_size; (void)ws_size;
  const int*   X     = (const int*)  d_in[0];
  const float* emb   = (const float*)d_in[1];
  const float* W_ih0 = (const float*)d_in[2];
  const float* W_hh0 = (const float*)d_in[3];
  const float* b_ih0 = (const float*)d_in[4];
  const float* b_hh0 = (const float*)d_in[5];
  const float* W_ih1 = (const float*)d_in[6];
  const float* W_hh1 = (const float*)d_in[7];
  const float* b_ih1 = (const float*)d_in[8];
  const float* b_hh1 = (const float*)d_in[9];
  const float* W_fc  = (const float*)d_in[10];
  const float* b_fc  = (const float*)d_in[11];

  // Workspace carve-up (~52.5 MB total)
  char* ws = (char*)d_ws;
  size_t off = 0;
  auto alloc = [&](size_t bytes) -> char* {
    char* p = ws + off;
    off += (bytes + 255) & ~(size_t)255;
    return p;
  };
  _Float16* Xe    = (_Float16*)alloc((size_t)BT * Ee * 2);
  _Float16* Wih0h = (_Float16*)alloc((size_t)Hh * Ee * 2);
  _Float16* Whh0h = (_Float16*)alloc((size_t)Hh * Hh * 2);
  _Float16* Wih1h = (_Float16*)alloc((size_t)Hh * Hh * 2);
  _Float16* Whh1h = (_Float16*)alloc((size_t)Hh * Hh * 2);
  _Float16* Wfch  = (_Float16*)alloc((size_t)Vv * Hh * 2);
  float*    xpbuf = (float*)   alloc((size_t)BT * Hh * 4);  // reused for xp0 & xp1
  _Float16* out1h = (_Float16*)alloc((size_t)BT * Hh * 2);
  _Float16* out2h = (_Float16*)alloc((size_t)BT * Hh * 2);

  // d_out layout: logits [B,T,V] | hidden [2,B,H] | out2 [B,T,H]
  float* logits = (float*)d_out;
  float* h1 = logits + (size_t)Bb * Tt * Vv;
  float* h2 = h1 + (size_t)Bb * Hh;
  float* out2f = h2 + (size_t)Bb * Hh;

  // Precision conversion (weights read once; activations gathered+converted)
  gather_embed_f16<<<BT, Ee, 0, stream>>>(X, emb, Xe);
  auto cvt = [&](const float* src, _Float16* dst, int n) {
    cvt_f32_to_f16<<<(n + 255) / 256, 256, 0, stream>>>(src, dst, n);
  };
  cvt(W_ih0, Wih0h, Hh * Ee);
  cvt(W_hh0, Whh0h, Hh * Hh);
  cvt(W_ih1, Wih1h, Hh * Hh);
  cvt(W_hh1, Whh1h, Hh * Hh);
  cvt(W_fc,  Wfch,  Vv * Hh);

  // Layer 0: xp0 = Xe @ W_ih0^T + b_ih0 + b_hh0 ; then sequential scan
  wmma_gemm_abt<<<dim3(Hh / 256, BT / 32), 256, 0, stream>>>(
      Xe, Wih0h, b_ih0, b_hh0, xpbuf, BT, Hh, Ee, 0);
  rnn_scan<<<1, 1024, 0, stream>>>(Whh0h, xpbuf, out1h, h1, nullptr);

  // Layer 1
  wmma_gemm_abt<<<dim3(Hh / 256, BT / 32), 256, 0, stream>>>(
      out1h, Wih1h, b_ih1, b_hh1, xpbuf, BT, Hh, Hh, 0);
  rnn_scan<<<1, 1024, 0, stream>>>(Whh1h, xpbuf, out2h, h2, out2f);

  // Classifier: logits = out2 @ W_fc^T + b_fc  (134 GFLOP, the hot GEMM)
  wmma_gemm_abt<<<dim3(Vv / 256, BT / 32), 256, 0, stream>>>(
      out2h, Wfch, b_fc, nullptr, logits, BT, Vv, Hh, 1);
}
